// ContextualAttention_46986942218432
// MI455X (gfx1250) — matrix operand order repaired and checked by reading, not compiled, not run on hardware
//
#include <hip/hip_runtime.h>

// Problem constants (from reference): B=4, S=1024, D=1024, H=16, DH=64
#define BB 4
#define SS 1024
#define DD 1024
#define HH 16
#define DHD 64
#define NROW (BB * SS)      // 4096 rows (b*S+s)
#define NQKV (3 * DD)       // 3072

typedef __attribute__((ext_vector_type(16))) __bf16 v16bf;
typedef __attribute__((ext_vector_type(8)))  float  v8f;

// ---------------------------------------------------------------------------
// WMMA fragment loaders (CDNA5 16-bit layouts, wave32)
// A 16x32 (MxK): lanes 0-15 -> M=lane, K-half 0; lanes 16-31 -> M=lane-16, K-half 8.
//   VGPR j (j<4): K = half+2j, half+2j+1 ; VGPR 4+j: K = 16+half+2j, +1
// B 32x16 (KxN): lanes 0-15 -> N=lane, K-half 0; lanes 16-31 -> N=lane-16, K-half 8.
//   VGPR j (j<4): K = half+2j, half+2j+1 ; VGPR 4+j: K = 16+half+2j, +1
// Both loaders read from a [row][k] layout (for B the tile is stored
// transposed, i.e. [n][k]), so every pair is one aligned 32-bit load.
// ---------------------------------------------------------------------------
static __device__ __forceinline__ v16bf frag_a_load(const __bf16* base, int ld) {
  int lane = threadIdx.x & 31;
  int r = lane & 15;
  int khalf = (lane >> 4) << 3;   // 0 or 8
  const __bf16* rp = base + r * ld;
  v16bf f;
#pragma unroll
  for (int j = 0; j < 4; ++j) {
    union { unsigned u; __bf16 h[2]; } w0, w1;
    w0.u = *reinterpret_cast<const unsigned*>(rp + khalf + 2 * j);
    w1.u = *reinterpret_cast<const unsigned*>(rp + 16 + khalf + 2 * j);
    f[2 * j]     = w0.h[0];  f[2 * j + 1] = w0.h[1];
    f[8 + 2 * j] = w1.h[0];  f[9 + 2 * j] = w1.h[1];
  }
  return f;
}

static __device__ __forceinline__ v16bf frag_b_load(const __bf16* base, int ld) {
  int lane = threadIdx.x & 31;
  int n = lane & 15;
  int koff = (lane >> 4) << 3;    // 0 or 8
  const __bf16* rp = base + n * ld + koff;
  v16bf f;
#pragma unroll
  for (int j = 0; j < 8; ++j) {
    union { unsigned u; __bf16 h[2]; } w;
    w.u = *reinterpret_cast<const unsigned*>(rp + 2 * j + (j >= 4 ? 8 : 0));
    f[2 * j] = w.h[0];  f[2 * j + 1] = w.h[1];
  }
  return f;
}

// ---------------------------------------------------------------------------
// fp32 -> bf16 conversion (2 elements/thread, packed dword store).
// All our element counts are multiples of 2.
// ---------------------------------------------------------------------------
__global__ void k_f32_to_bf16(const float* __restrict__ src,
                              __bf16* __restrict__ dst, int n2) {
  int i = blockIdx.x * blockDim.x + threadIdx.x;   // pair index
  if (i < n2) {
    float a = src[2 * i], b = src[2 * i + 1];
    union { unsigned u; __bf16 h[2]; } w;
    w.h[0] = (__bf16)a;  w.h[1] = (__bf16)b;
    *reinterpret_cast<unsigned*>(dst + 2 * i) = w.u;
  }
}

// ---------------------------------------------------------------------------
// Tiled bf16 GEMM, fp32 accumulate: D = A[M,K] @ B[K,N] + bias[N]
// Block tile 128(M) x 128(N), K-step 32, 256 threads = 8 waves (4x2 wave grid).
// Each wave computes a 32x64 tile as 2x4 WMMA fragments (8 WMMA / K-step).
// MODE 0: write fp32 to C.
// MODE 1: fused QKV epilogue -> per-head bf16 [B,H,S,DH] into Qo/Ko/Vo,
//         with 1/sqrt(DH) folded into Q.
// ---------------------------------------------------------------------------
template <int MODE>
__global__ __launch_bounds__(256)
void k_gemm_bf16(const __bf16* __restrict__ A, const __bf16* __restrict__ Bm,
                 const float* __restrict__ bias, float* __restrict__ C,
                 __bf16* __restrict__ Qo, __bf16* __restrict__ Ko,
                 __bf16* __restrict__ Vo, int M, int N, int K) {
  const int LDA = 34;   // 17-bank stride -> conflict free
  const int LDB = 34;
  __shared__ __bf16 sA[128 * 34];
  __shared__ __bf16 sB[128 * 34];   // stored transposed: sB[n][k]

  int m0 = blockIdx.y * 128;
  int n0 = blockIdx.x * 128;
  int tid = threadIdx.x;
  int wave = tid >> 5;
  int wm = wave >> 1;   // 0..3 -> M offset 32*wm
  int wn = wave & 1;    // 0..1 -> N offset 64*wn
  int lane = tid & 31;

  v8f acc[2][4];
#pragma unroll
  for (int a = 0; a < 2; ++a)
#pragma unroll
    for (int b = 0; b < 4; ++b)
#pragma unroll
      for (int e = 0; e < 8; ++e) acc[a][b][e] = 0.0f;

  for (int k0 = 0; k0 < K; k0 += 32) {
    // A tile: 128x32 bf16 via packed dword loads (2048 dwords / 256 threads)
#pragma unroll
    for (int i = 0; i < 8; ++i) {
      int idx = tid + i * 256;
      int row = idx >> 4;
      int cp = idx & 15;
      unsigned w = *reinterpret_cast<const unsigned*>(
          A + (long)(m0 + row) * K + k0 + cp * 2);
      *reinterpret_cast<unsigned*>(&sA[row * LDA + cp * 2]) = w;
    }
    // B tile transposed into LDS: sB[n][k] = B[k0+k][n0+n]  (coalesced on n)
#pragma unroll
    for (int i = 0; i < 16; ++i) {
      int idx = tid + i * 256;
      int n = idx & 127;
      int k = idx >> 7;
      sB[n * LDB + k] = Bm[(long)(k0 + k) * N + n0 + n];
    }
    // prefetch next K-tile (lowers to global_prefetch_b8)
    if (k0 + 32 < K) {
      __builtin_prefetch(A + (long)(m0 + (tid >> 1)) * K + k0 + 32, 0, 1);
      __builtin_prefetch(Bm + (long)(k0 + 32 + (tid >> 7)) * N + n0 + (tid & 127), 0, 1);
    }
    __syncthreads();

    v16bf af[2];
#pragma unroll
    for (int fm = 0; fm < 2; ++fm)
      af[fm] = frag_a_load(&sA[(wm * 32 + fm * 16) * LDA], LDA);
#pragma unroll
    for (int fn = 0; fn < 4; ++fn) {
      v16bf bfr = frag_b_load(&sB[(wn * 64 + fn * 16) * LDB], LDB);
#pragma unroll
      for (int fm = 0; fm < 2; ++fm)
        acc[fm][fn] = __builtin_amdgcn_wmma_f32_16x16x32_bf16(
            false, af[fm], false, bfr, (short)0, acc[fm][fn], false, false);
    }
    __syncthreads();
  }

  // Epilogue. C layout: VGPR r, lanes 0-15: (M=r, N=lane); lanes 16-31: (M=r+8)
  int rl = (lane >> 4) << 3;
  int cl = lane & 15;
#pragma unroll
  for (int fm = 0; fm < 2; ++fm)
#pragma unroll
    for (int fn = 0; fn < 4; ++fn)
#pragma unroll
      for (int r = 0; r < 8; ++r) {
        int row = m0 + wm * 32 + fm * 16 + r + rl;
        int col = n0 + wn * 64 + fn * 16 + cl;
        float val = acc[fm][fn][r] + bias[col];
        if (MODE == 0) {
          C[(long)row * N + col] = val;
        } else {
          // fused QKV scatter: row = b*S+s ; col -> (which, h, dh)
          int which = col >> 10;
          int rem = col & 1023;
          int h = rem >> 6, dh = rem & 63;
          int b = row >> 10, s = row & 1023;
          long o = (((long)(b * HH + h)) * SS + s) * DHD + dh;
          if (which == 0)      Qo[o] = (__bf16)(val * 0.125f);  // 1/sqrt(64)
          else if (which == 1) Ko[o] = (__bf16)val;
          else                 Vo[o] = (__bf16)val;
        }
      }
}

// ---------------------------------------------------------------------------
// Flash-style causal attention. Grid: (S/64 q-tiles, B*H). Block: 128 = 4 waves.
// Each wave owns 16 Q rows; loops over causal KV tiles of 64.
// Output written bf16, heads merged: [b*S+s][h*64+dh].
// ---------------------------------------------------------------------------
__global__ __launch_bounds__(128)
void k_attn(const __bf16* __restrict__ Q, const __bf16* __restrict__ Kg,
            const __bf16* __restrict__ V, __bf16* __restrict__ O) {
  const int LK = 72, LV = 72, LP = 72;
  __shared__ __bf16 sK[64 * 72];          // [kv][dh]  (== B operand [n][k] for QK^T)
  __shared__ __bf16 sV[64 * 72];          // [dh][kv]  (== B operand [n][k] for P@V)
  __shared__ __bf16 sP[4 * 16 * 72];      // per-wave P strips

  int qt = blockIdx.x;                    // 0..15
  int bh = blockIdx.y;                    // 0..63
  int tid = threadIdx.x;
  int wave = tid >> 5;
  int lane = tid & 31;
  long base = (long)bh * SS * DHD;
  int q0 = qt * 64;
  int qrow0 = q0 + wave * 16;
  int rl = (lane >> 4) << 3;
  int cl = lane & 15;

  // Q fragments (pre-scaled by 1/sqrt(DH)); loop-invariant
  v16bf qf[2];
  qf[0] = frag_a_load(Q + base + (long)qrow0 * DHD, DHD);
  qf[1] = frag_a_load(Q + base + (long)qrow0 * DHD + 32, DHD);

  v8f o[4];
#pragma unroll
  for (int fo = 0; fo < 4; ++fo)
#pragma unroll
    for (int e = 0; e < 8; ++e) o[fo][e] = 0.0f;
  float mrow[8], lrow[8];
#pragma unroll
  for (int r = 0; r < 8; ++r) { mrow[r] = -__builtin_inff(); lrow[r] = 0.0f; }

  for (int t = 0; t <= qt; ++t) {
    int kv0 = t * 64;
    // K tile row-major into LDS (packed dword loads: 2048 dwords / 128 threads)
#pragma unroll
    for (int i = 0; i < 16; ++i) {
      int idx = tid + i * 128;
      int row = idx >> 5, cp = idx & 31;
      unsigned w = *reinterpret_cast<const unsigned*>(
          Kg + base + (long)(kv0 + row) * DHD + cp * 2);
      *reinterpret_cast<unsigned*>(&sK[row * LK + cp * 2]) = w;
    }
    // V tile transposed into LDS: sV[d][kv]
#pragma unroll
    for (int i = 0; i < 32; ++i) {
      int idx = tid + i * 128;
      int d = idx & 63, row = idx >> 6;
      sV[d * LV + row] = V[base + (long)(kv0 + row) * DHD + d];
    }
    __syncthreads();

    // S = Q @ K^T   (16x64 per wave, 8 WMMA)
    v8f sc[4];
#pragma unroll
    for (int fn = 0; fn < 4; ++fn)
#pragma unroll
      for (int e = 0; e < 8; ++e) sc[fn][e] = 0.0f;
#pragma unroll
    for (int ks = 0; ks < 2; ++ks)
#pragma unroll
      for (int fn = 0; fn < 4; ++fn) {
        v16bf bb = frag_b_load(&sK[(fn * 16) * LK + ks * 32], LK);
        sc[fn] = __builtin_amdgcn_wmma_f32_16x16x32_bf16(
            false, qf[ks], false, bb, (short)0, sc[fn], false, false);
      }

    // causal mask in C layout
#pragma unroll
    for (int fn = 0; fn < 4; ++fn)
#pragma unroll
      for (int r = 0; r < 8; ++r) {
        int mr = qrow0 + r + rl;
        int nc = kv0 + fn * 16 + cl;
        if (nc > mr) sc[fn][r] = -__builtin_inff();
      }

    // online softmax per row (rows replicated across 16-lane halves)
#pragma unroll
    for (int r = 0; r < 8; ++r) {
      float mx = fmaxf(fmaxf(sc[0][r], sc[1][r]), fmaxf(sc[2][r], sc[3][r]));
#pragma unroll
      for (int m = 1; m <= 8; m <<= 1) mx = fmaxf(mx, __shfl_xor(mx, m, 32));
      float mnew = fmaxf(mrow[r], mx);
      float scale = __expf(mrow[r] - mnew);
      float sum = 0.0f;
#pragma unroll
      for (int fn = 0; fn < 4; ++fn) {
        float p = __expf(sc[fn][r] - mnew);
        sc[fn][r] = p;
        sum += p;
      }
#pragma unroll
      for (int m = 1; m <= 8; m <<= 1) sum += __shfl_xor(sum, m, 32);
      lrow[r] = lrow[r] * scale + sum;
      mrow[r] = mnew;
#pragma unroll
      for (int fo = 0; fo < 4; ++fo) o[fo][r] *= scale;
    }

    // P -> LDS bf16 (re-enter A-fragment layout)
    __bf16* pw = &sP[wave * 16 * LP];
#pragma unroll
    for (int fn = 0; fn < 4; ++fn)
#pragma unroll
      for (int r = 0; r < 8; ++r)
        pw[(r + rl) * LP + fn * 16 + cl] = (__bf16)sc[fn][r];

    // O += P @ V  (8 WMMA)
#pragma unroll
    for (int ks = 0; ks < 2; ++ks) {
      v16bf pa = frag_a_load(pw + ks * 32, LP);
#pragma unroll
      for (int fo = 0; fo < 4; ++fo) {
        v16bf bb = frag_b_load(&sV[(fo * 16) * LV + ks * 32], LV);
        o[fo] = __builtin_amdgcn_wmma_f32_16x16x32_bf16(
            false, pa, false, bb, (short)0, o[fo], false, false);
      }
    }
    __syncthreads();
  }

  // epilogue: normalize by l, merge heads, write bf16
  int h = bh & (HH - 1);
  int b = bh >> 4;
#pragma unroll
  for (int fo = 0; fo < 4; ++fo)
#pragma unroll
    for (int r = 0; r < 8; ++r) {
      int s = q0 + wave * 16 + r + rl;
      int d = h * DHD + fo * 16 + cl;
      long row = (long)b * SS + s;
      O[row * DD + d] = (__bf16)(o[fo][r] / lrow[r]);
    }
}

// ---------------------------------------------------------------------------
// Host-side orchestration
// ---------------------------------------------------------------------------
extern "C" void kernel_launch(void* const* d_in, const int* in_sizes, int n_in,
                              void* d_out, int out_size, void* d_ws, size_t ws_size,
                              hipStream_t stream) {
  const float* x        = (const float*)d_in[0];
  const float* c_attn_w = (const float*)d_in[1];
  const float* c_attn_b = (const float*)d_in[2];
  const float* c_proj_w = (const float*)d_in[3];
  const float* c_proj_b = (const float*)d_in[4];
  float* out = (float*)d_out;

  // Workspace carve-out (~50 MB total)
  char* p = (char*)d_ws;
  __bf16* xb    = (__bf16*)p; p += (size_t)NROW * DD * 2;        // 8.39 MB
  __bf16* wqkv  = (__bf16*)p; p += (size_t)DD * NQKV * 2;        // 6.29 MB
  __bf16* wproj = (__bf16*)p; p += (size_t)DD * DD * 2;          // 2.10 MB
  __bf16* qb    = (__bf16*)p; p += (size_t)NROW * DD * 2;        // 8.39 MB
  __bf16* kb    = (__bf16*)p; p += (size_t)NROW * DD * 2;        // 8.39 MB
  __bf16* vb    = (__bf16*)p; p += (size_t)NROW * DD * 2;        // 8.39 MB
  __bf16* ab    = (__bf16*)p; p += (size_t)NROW * DD * 2;        // 8.39 MB

  // 1) fp32 -> bf16 conversions (pairwise)
  {
    int n2 = (NROW * DD) / 2;
    k_f32_to_bf16<<<(n2 + 255) / 256, 256, 0, stream>>>(x, xb, n2);
  }
  {
    int n2 = (DD * NQKV) / 2;
    k_f32_to_bf16<<<(n2 + 255) / 256, 256, 0, stream>>>(c_attn_w, wqkv, n2);
  }
  {
    int n2 = (DD * DD) / 2;
    k_f32_to_bf16<<<(n2 + 255) / 256, 256, 0, stream>>>(c_proj_w, wproj, n2);
  }

  // 2) QKV GEMM with fused bias + per-head scatter + Q scaling (bf16 out)
  k_gemm_bf16<1><<<dim3(NQKV / 128, NROW / 128), 256, 0, stream>>>(
      xb, wqkv, c_attn_b, nullptr, qb, kb, vb, NROW, NQKV, DD);

  // 3) flash attention
  k_attn<<<dim3(SS / 64, BB * HH), 128, 0, stream>>>(qb, kb, vb, ab);

  // 4) output projection: [4096,1024] = ab @ wproj + c_proj_b -> d_out (fp32)
  k_gemm_bf16<0><<<dim3(DD / 128, NROW / 128), 256, 0, stream>>>(
      ab, wproj, c_proj_b, out, nullptr, nullptr, nullptr, NROW, DD, DD);
}
